// LinearAttention_22273700397440
// MI455X (gfx1250) — compile-verified, matrix operand order checked by reading, not run on hardware
//
#include <hip/hip_runtime.h>
#include <hip/hip_bf16.h>
#include <cstdint>

#define DIM   1024
#define NHEAD 16
#define HDIM  64
#define BATCH 4
#define SEQ   8192
#define MROWS (BATCH*SEQ)          // 32768
#define QK_SCALE 0.125f            // HDIM^-0.5

// GEMM tiling (big GEMMs)
#define BM 128
#define BN2 128                    // wide N tile to cut L2 re-reads
#define BK 64
#define LDA 72                     // bf16 elems; 144B rows (16B aligned, conflict-free)
#define LDC2 132                   // f32 elems for C spill
#define TILE_ELEMS (BM * LDA)      // 9216 bf16 = 18432 B per buffer
#define NSTEP (DIM / BK)           // 16

// small GEMM tiling (k2/k3a)
#define BN 64
#define LDB 72
#define NSPLIT 16
#define CHUNK (SEQ / NSPLIT)       // 512
#define LDE 40                     // 80B rows

typedef __bf16 bf16;
typedef bf16  v16bf __attribute__((ext_vector_type(16)));
typedef bf16  v8bf  __attribute__((ext_vector_type(8)));
typedef bf16  v4bf  __attribute__((ext_vector_type(4)));
typedef float v8f   __attribute__((ext_vector_type(8)));
typedef unsigned int v4u __attribute__((ext_vector_type(4)));
typedef int   v8i   __attribute__((ext_vector_type(8)));
typedef int   v4i   __attribute__((ext_vector_type(4)));

__device__ __forceinline__ bf16 f2bf(float f) {
  uint32_t u = __builtin_bit_cast(uint32_t, f);
  u = (u + 0x7FFFu + ((u >> 16) & 1u)) >> 16;     // round-to-nearest-even
  uint16_t s = (uint16_t)u;
  return __builtin_bit_cast(bf16, s);
}
__device__ __forceinline__ float bf2f(bf16 b) {
  uint32_t u = ((uint32_t)__builtin_bit_cast(uint16_t, b)) << 16;
  return __builtin_bit_cast(float, u);
}

// LDS byte offset of a generic pointer into shared memory (addrspace(3) ptrtoint)
__device__ __forceinline__ uint32_t lds_off(const void* p) {
  return (uint32_t)(uintptr_t)(__attribute__((address_space(3))) const void*)p;
}

// Issue one TDM 2-D tile load: tile_d1 rows x tile_d0 bf16 elems from a
// row-major tensor (row stride stride0 elems) into LDS, padding +4 dwords
// every 32 dwords so each 64-elem row lands at a 72-elem (144 B) stride.
__device__ __forceinline__ void tdm_load_2d(uint32_t lds_byte_off, const bf16* gptr,
                                            uint32_t tensor_d0, uint32_t tensor_d1,
                                            uint32_t tile_d0, uint32_t tile_d1,
                                            uint32_t stride0) {
  const uint64_t ga = (uint64_t)(uintptr_t)gptr;
  v4u g0;
  g0[0] = 1u;                                     // count=1, user descriptor
  g0[1] = lds_byte_off;                           // lds_addr
  g0[2] = (uint32_t)ga;                           // global_addr[31:0]
  g0[3] = (uint32_t)(ga >> 32) | (2u << 30);      // global_addr[56:32] | type=2
  v8i g1;
  g1[0] = (int)((1u << 16)        // data_size = 1 -> 2 bytes
              | (1u << 20)        // pad_enable
              | (4u << 22)        // pad_interval: 4 -> every 32 dwords (64 bf16)
              | (3u << 25));      // pad_amount:   3 -> 4 dwords (8 bf16)
  g1[1] = (int)(tensor_d0 << 16);                           // tensor_dim0[15:0]
  g1[2] = (int)((tensor_d0 >> 16) | (tensor_d1 << 16));     // d0 hi | d1 lo
  g1[3] = (int)((tensor_d1 >> 16) | (tile_d0 << 16));       // d1 hi | tile_dim0
  g1[4] = (int)(tile_d1);                                   // tile_dim1 (tile_dim2=0)
  g1[5] = (int)stride0;                                     // tensor_dim0_stride lo
  g1[6] = 0;
  g1[7] = 0;
  v4i z4; z4[0] = 0; z4[1] = 0; z4[2] = 0; z4[3] = 0;
  v8i z8;
  #pragma unroll
  for (int i = 0; i < 8; ++i) z8[i] = 0;
  __builtin_amdgcn_tensor_load_to_lds(g0, g1, z4, z4, z8, 0);
}

// Load one 16(M/N) x 32(K) WMMA A/B fragment from an LDS tile laid out
// [row][k], leading dimension ld elems (row byte size 16B aligned).
// lanes 0-15 : row = lane,    K = {k0..k0+7} and {k0+16..k0+23}
// lanes 16-31: row = lane-16, K = {k0+8..k0+15} and {k0+24..k0+31}
__device__ __forceinline__ v16bf frag_ld(const bf16* tile, int ld, int row0, int k0) {
  const int lane = threadIdx.x & 31;
  const bf16* p = tile + (row0 + (lane & 15)) * ld + k0 + ((lane >> 4) << 3);
  union { v16bf v; v8bf h[2]; } f;
  f.h[0] = *(const v8bf*)(p);
  f.h[1] = *(const v8bf*)(p + 16);
  return f.v;
}

__device__ __forceinline__ v8f wmma_bf16(v16bf a, v16bf b, v8f c) {
  return __builtin_amdgcn_wmma_f32_16x16x32_bf16(false, a, false, b,
                                                 (short)0, c, false, false);
}

extern __shared__ char smem[];

// ---------------------------------------------------------------- kernel 0
// bf16 conversions + zeroing of atomic accumulators (every launch).
__global__ void k0_init(const float* __restrict__ x, const float* __restrict__ wqkv,
                        const float* __restrict__ wproj,
                        bf16* __restrict__ x_bf, bf16* __restrict__ wqkv_bf,
                        bf16* __restrict__ wproj_bf,
                        float* __restrict__ ctx, float* __restrict__ zbuf) {
  const int i = blockIdx.x * blockDim.x + threadIdx.x;
  if (i < MROWS * DIM)   x_bf[i] = f2bf(x[i]);
  if (i < 3 * DIM * DIM) wqkv_bf[i] = f2bf(wqkv[i]);
  if (i < DIM * DIM)     wproj_bf[i] = f2bf(wproj[i]);
  if (i < BATCH * NHEAD * HDIM * HDIM) ctx[i] = 0.f;
  if (i < BATCH * NHEAD * HDIM)        zbuf[i] = 0.f;
}

// ------------------------------------------------------ kernel 1: QKV GEMM
// qkv = x @ w_qkv^T  (TDM double-buffered into LDS, bf16 WMMA), fused
// epilogue per 128-column (two-head-aligned) slab.
__global__ void __launch_bounds__(256)
k1_qkv(const bf16* __restrict__ xbf, const bf16* __restrict__ wbf,
       bf16* __restrict__ qs, bf16* __restrict__ ebuf, bf16* __restrict__ vbuf) {
  bf16*  As   = (bf16*)smem;                        // 2 x 18432 B
  bf16*  Bs   = (bf16*)(smem + 2 * 18432);          // 2 x 18432 B
  float* Cs   = (float*)(smem + 4 * 18432);         // 128*132*4 = 67584 B
  float* rmax = (float*)(smem + 4 * 18432 + 67584); // 2*128 f32
  float* rscl = rmax + 2 * BM;

  const int tid  = threadIdx.x;
  const int wave = tid >> 5;
  const int mb   = blockIdx.x * BM;
  const int cb   = blockIdx.y * BN2;                // 0..2944

  v8f acc[8];
  #pragma unroll
  for (int t = 0; t < 8; ++t)
    #pragma unroll
    for (int i = 0; i < 8; ++i) acc[t][i] = 0.f;

  const uint32_t aoff = lds_off(As);
  const uint32_t boff = lds_off(Bs);

  if (tid < 32) {  // wave 0 drives the TDM pipeline
    tdm_load_2d(aoff, xbf + (size_t)mb * DIM, DIM, MROWS, BK, BM, DIM);
    tdm_load_2d(boff, wbf + (size_t)cb * DIM, DIM, 3 * DIM, BK, BN2, DIM);
  }

  for (int s = 0; s < NSTEP; ++s) {
    if (tid < 32) {
      if (s + 1 < NSTEP) {
        const int nb = (s + 1) & 1;
        const int kb = (s + 1) * BK;
        tdm_load_2d(aoff + nb * 18432u, xbf + (size_t)mb * DIM + kb,
                    DIM, MROWS, BK, BM, DIM);
        tdm_load_2d(boff + nb * 18432u, wbf + (size_t)cb * DIM + kb,
                    DIM, 3 * DIM, BK, BN2, DIM);
        __builtin_amdgcn_s_wait_tensorcnt(2);   // stage s complete (in-order)
      } else {
        __builtin_amdgcn_s_wait_tensorcnt(0);
      }
    }
    __syncthreads();
    const bf16* Ab = As + (s & 1) * TILE_ELEMS;
    const bf16* Bb = Bs + (s & 1) * TILE_ELEMS;
    const int wr = wave * 16;
    #pragma unroll
    for (int kc = 0; kc < BK; kc += 32) {
      const v16bf va = frag_ld(Ab, LDA, wr, kc);
      #pragma unroll
      for (int t = 0; t < 8; ++t) {
        const v16bf vbr = frag_ld(Bb, LDA, t * 16, kc);
        acc[t] = wmma_bf16(va, vbr, acc[t]);
      }
    }
    __syncthreads();
  }

  // spill accumulators (C layout: lane = n + 16*(m>=8), vgpr = m%8)
  {
    const int lane = tid & 31;
    const int rb = wave * 16 + ((lane >> 4) << 3);
    const int cl = lane & 15;
    #pragma unroll
    for (int t = 0; t < 8; ++t)
      #pragma unroll
      for (int i = 0; i < 8; ++i)
        Cs[(rb + i) * LDC2 + t * 16 + cl] = acc[t][i];
  }
  __syncthreads();

  const int sec = cb >> 10;                    // 0=q, 1=k, 2=v
  const int ci  = cb & 1023;
  if (sec == 0) {
    // row softmax per 64-wide head (two heads per slab), * SCALE
    {
      const int r = tid & 127, hh = tid >> 7;
      const float* row = Cs + r * LDC2 + hh * 64;
      float m = -1e30f, sum = 0.f;
      for (int e = 0; e < 64; ++e) m = fmaxf(m, row[e]);
      for (int e = 0; e < 64; ++e) sum += __expf(row[e] - m);
      rmax[hh * BM + r] = m;
      rscl[hh * BM + r] = QK_SCALE / sum;
    }
    __syncthreads();
    for (int i = tid; i < BM * BN2; i += 256) {
      const int r = i >> 7, e = i & 127, hh = e >> 6;
      const float v = __expf(Cs[r * LDC2 + e] - rmax[hh * BM + r]) * rscl[hh * BM + r];
      qs[(size_t)(mb + r) * DIM + ci + e] = f2bf(v);
    }
  } else if (sec == 1) {
    // E = exp(k_raw); k values ~N(0,1) so no max subtraction needed
    for (int i = tid; i < BM * BN2; i += 256) {
      const int r = i >> 7, e = i & 127;
      ebuf[(size_t)(mb + r) * DIM + ci + e] = f2bf(__expf(Cs[r * LDC2 + e]));
    }
  } else {
    for (int i = tid; i < BM * BN2; i += 256) {
      const int r = i >> 7, e = i & 127;
      vbuf[(size_t)(mb + r) * DIM + ci + e] = f2bf(Cs[r * LDC2 + e]);
    }
  }
}

// ---------------------------------- kernel 2: context' = E^T V, Z = colsum(E)
__global__ void __launch_bounds__(128)
k2_context(const bf16* __restrict__ ebuf, const bf16* __restrict__ vbuf,
           float* __restrict__ ctx, float* __restrict__ zbuf) {
  __shared__ bf16 Es[HDIM * LDE];              // [d][n] (transposed)
  __shared__ bf16 Vs[HDIM * LDE];              // [e][n] (transposed)
  const int tid  = threadIdx.x;
  const int wave = tid >> 5;
  const int bh = blockIdx.x >> 4;              // NSPLIT == 16
  const int ns = blockIdx.x & (NSPLIT - 1);
  const int b = bh >> 4, h = bh & 15;
  const size_t rowbase = (size_t)b * SEQ + (size_t)ns * CHUNK;
  const int hoff = h * HDIM;

  v8f acc[4];
  #pragma unroll
  for (int t = 0; t < 4; ++t)
    #pragma unroll
    for (int i = 0; i < 8; ++i) acc[t][i] = 0.f;
  float zacc = 0.f;

  for (int nb = 0; nb < CHUNK; nb += 32) {
    for (int i = tid; i < 32 * HDIM; i += 128) {   // transpose into LDS
      const int nl = i >> 6, d = i & 63;
      Es[d * LDE + nl] = ebuf[(rowbase + nb + nl) * DIM + hoff + d];
      Vs[d * LDE + nl] = vbuf[(rowbase + nb + nl) * DIM + hoff + d];
    }
    __syncthreads();
    if (tid < HDIM) {
      #pragma unroll
      for (int n = 0; n < 32; ++n) zacc += bf2f(Es[tid * LDE + n]);
    }
    const int wr = wave * 16;
    const v16bf va = frag_ld(Es, LDE, wr, 0);
    #pragma unroll
    for (int t = 0; t < 4; ++t) {
      const v16bf vbr = frag_ld(Vs, LDE, t * 16, 0);
      acc[t] = wmma_bf16(va, vbr, acc[t]);
    }
    __syncthreads();
  }
  {
    const int lane = tid & 31;
    const int rb = wave * 16 + ((lane >> 4) << 3);
    const int cl = lane & 15;
    float* cbase = ctx + (size_t)bh * HDIM * HDIM;
    #pragma unroll
    for (int t = 0; t < 4; ++t)
      #pragma unroll
      for (int i = 0; i < 8; ++i)
        atomicAdd(cbase + (rb + i) * HDIM + t * 16 + cl, acc[t][i]);
  }
  if (tid < HDIM) atomicAdd(zbuf + bh * HDIM + tid, zacc);
}

// ------------------- kernel 2b: ctxT[bh][e][d] = ctx[bh][d][e] / Z[bh][d]
__global__ void k2b_norm(const float* __restrict__ ctx, const float* __restrict__ zbuf,
                         bf16* __restrict__ ctxT) {
  const int i = blockIdx.x * blockDim.x + threadIdx.x;
  if (i >= BATCH * NHEAD * HDIM * HDIM) return;
  const int bh = i >> 12, e = (i >> 6) & 63, d = i & 63;
  ctxT[i] = f2bf(ctx[(size_t)bh * 4096 + d * 64 + e] / zbuf[bh * 64 + d]);
}

// --------------------------- kernel 3a: t = q_soft @ ctx (block-diag/head)
__global__ void __launch_bounds__(256)
k3a_qctx(const bf16* __restrict__ qs, const bf16* __restrict__ ctxT,
         bf16* __restrict__ tbuf) {
  __shared__ bf16 As3[BM * LDA];
  __shared__ bf16 Bs3[BN * LDB];
  const int tid = threadIdx.x, wave = tid >> 5;
  const int mb = blockIdx.x * BM;
  const int h = blockIdx.y;
  const int bh = (mb / SEQ) * NHEAD + h;
  const int hoff = h * HDIM;

  for (int i = tid; i < BM * (HDIM / 4); i += 256) {
    const int r = i >> 4, c4 = (i & 15) * 4;
    *(v4bf*)(As3 + r * LDA + c4) =
        *(const v4bf*)(qs + (size_t)(mb + r) * DIM + hoff + c4);
  }
  for (int i = tid; i < BN * (HDIM / 4); i += 256) {
    const int r = i >> 4, c4 = (i & 15) * 4;
    *(v4bf*)(Bs3 + r * LDB + c4) =
        *(const v4bf*)(ctxT + (size_t)bh * 4096 + r * HDIM + c4);
  }
  __syncthreads();

  v8f acc[4];
  #pragma unroll
  for (int t = 0; t < 4; ++t)
    #pragma unroll
    for (int i = 0; i < 8; ++i) acc[t][i] = 0.f;

  const int wr = wave * 16;
  #pragma unroll
  for (int kc = 0; kc < HDIM; kc += 32) {
    const v16bf va = frag_ld(As3, LDA, wr, kc);
    #pragma unroll
    for (int t = 0; t < 4; ++t) {
      const v16bf vbr = frag_ld(Bs3, LDB, t * 16, kc);
      acc[t] = wmma_bf16(va, vbr, acc[t]);
    }
  }
  const int lane = tid & 31;
  const int rb = wr + ((lane >> 4) << 3);
  const int cl = lane & 15;
  #pragma unroll
  for (int t = 0; t < 4; ++t)
    #pragma unroll
    for (int i = 0; i < 8; ++i)
      tbuf[(size_t)(mb + rb + i) * DIM + hoff + t * 16 + cl] = f2bf(acc[t][i]);
}

// ----------------------------- kernel 3b: out = t @ w_proj^T + b_proj (f32)
__global__ void __launch_bounds__(256)
k3b_proj(const bf16* __restrict__ tbuf, const bf16* __restrict__ wpbf,
         const float* __restrict__ bias, float* __restrict__ out) {
  bf16*  As   = (bf16*)smem;                        // 2 x 18432 B
  bf16*  Bs   = (bf16*)(smem + 2 * 18432);          // 2 x 18432 B
  float* Cs   = (float*)(smem + 4 * 18432);         // 67584 B
  const int tid = threadIdx.x, wave = tid >> 5;
  const int mb = blockIdx.x * BM;
  const int cb = blockIdx.y * BN2;

  v8f acc[8];
  #pragma unroll
  for (int t = 0; t < 8; ++t)
    #pragma unroll
    for (int i = 0; i < 8; ++i) acc[t][i] = 0.f;

  const uint32_t aoff = lds_off(As);
  const uint32_t boff = lds_off(Bs);
  if (tid < 32) {
    tdm_load_2d(aoff, tbuf + (size_t)mb * DIM, DIM, MROWS, BK, BM, DIM);
    tdm_load_2d(boff, wpbf + (size_t)cb * DIM, DIM, DIM, BK, BN2, DIM);
  }

  for (int s = 0; s < NSTEP; ++s) {
    if (tid < 32) {
      if (s + 1 < NSTEP) {
        const int nb = (s + 1) & 1;
        const int kb = (s + 1) * BK;
        tdm_load_2d(aoff + nb * 18432u, tbuf + (size_t)mb * DIM + kb,
                    DIM, MROWS, BK, BM, DIM);
        tdm_load_2d(boff + nb * 18432u, wpbf + (size_t)cb * DIM + kb,
                    DIM, DIM, BK, BN2, DIM);
        __builtin_amdgcn_s_wait_tensorcnt(2);
      } else {
        __builtin_amdgcn_s_wait_tensorcnt(0);
      }
    }
    __syncthreads();
    const bf16* Ab = As + (s & 1) * TILE_ELEMS;
    const bf16* Bb = Bs + (s & 1) * TILE_ELEMS;
    const int wr = wave * 16;
    #pragma unroll
    for (int kc = 0; kc < BK; kc += 32) {
      const v16bf va = frag_ld(Ab, LDA, wr, kc);
      #pragma unroll
      for (int t = 0; t < 8; ++t) {
        const v16bf vbr = frag_ld(Bb, LDA, t * 16, kc);
        acc[t] = wmma_bf16(va, vbr, acc[t]);
      }
    }
    __syncthreads();
  }
  {
    const int lane = tid & 31;
    const int rb = wave * 16 + ((lane >> 4) << 3);
    const int cl = lane & 15;
    #pragma unroll
    for (int t = 0; t < 8; ++t)
      #pragma unroll
      for (int i = 0; i < 8; ++i)
        Cs[(rb + i) * LDC2 + t * 16 + cl] = acc[t][i];
  }
  __syncthreads();
  for (int i = tid; i < BM * BN2; i += 256) {
    const int r = i >> 7, c = i & 127;
    out[(size_t)(mb + r) * DIM + cb + c] = Cs[r * LDC2 + c] + bias[cb + c];
  }
}

// ---------------------------------------------------------------- launcher
extern "C" void kernel_launch(void* const* d_in, const int* in_sizes, int n_in,
                              void* d_out, int out_size, void* d_ws, size_t ws_size,
                              hipStream_t stream) {
  (void)in_sizes; (void)n_in; (void)out_size; (void)ws_size;
  const float* x      = (const float*)d_in[0];
  const float* w_qkv  = (const float*)d_in[1];
  const float* w_proj = (const float*)d_in[2];
  const float* b_proj = (const float*)d_in[3];
  float* out = (float*)d_out;

  char* p = (char*)d_ws;
  size_t off = 0;
  auto take = [&](size_t bytes) -> char* {
    char* r = p + off;
    off += (bytes + 255) & ~(size_t)255;
    return r;
  };
  bf16*  x_bf     = (bf16*)take((size_t)MROWS * DIM * 2);   // 64 MB
  bf16*  qs       = (bf16*)take((size_t)MROWS * DIM * 2);   // 64 MB
  bf16*  ebuf     = (bf16*)take((size_t)MROWS * DIM * 2);   // 64 MB (reused as t)
  bf16*  vbuf     = (bf16*)take((size_t)MROWS * DIM * 2);   // 64 MB
  bf16*  wqkv_bf  = (bf16*)take((size_t)3 * DIM * DIM * 2); // 6 MB
  bf16*  wproj_bf = (bf16*)take((size_t)DIM * DIM * 2);     // 2 MB
  float* ctx      = (float*)take((size_t)BATCH * NHEAD * HDIM * HDIM * 4);
  float* zbuf     = (float*)take((size_t)BATCH * NHEAD * HDIM * 4);
  bf16*  ctxT     = (bf16*)take((size_t)BATCH * NHEAD * HDIM * HDIM * 2);

  const size_t big_lds   = 4 * 18432 + 67584 + 2 * 2 * BM * 4;  // k1: ~143 KB
  const size_t proj_lds  = 4 * 18432 + 67584;                   // k3b: ~138 KB

  k0_init<<<(MROWS * DIM) / 256, 256, 0, stream>>>(x, w_qkv, w_proj, x_bf,
                                                   wqkv_bf, wproj_bf, ctx, zbuf);
  k1_qkv<<<dim3(MROWS / BM, (3 * DIM) / BN2), 256, big_lds, stream>>>(
      x_bf, wqkv_bf, qs, ebuf, vbuf);
  k2_context<<<BATCH * NHEAD * NSPLIT, 128, 0, stream>>>(ebuf, vbuf, ctx, zbuf);
  k2b_norm<<<(BATCH * NHEAD * HDIM * HDIM + 255) / 256, 256, 0, stream>>>(ctx, zbuf, ctxT);
  bf16* tbuf = ebuf;  // E is dead after k2 — reuse for t
  k3a_qctx<<<dim3(MROWS / BM, NHEAD), 256, 0, stream>>>(qs, ctxT, tbuf);
  k3b_proj<<<dim3(MROWS / BM, DIM / BN2), 256, proj_lds, stream>>>(
      tbuf, wproj_bf, b_proj, out);
}
